// TimeAwareFullAttention_22333829939818
// MI455X (gfx1250) — matrix operand order repaired and checked by reading, not compiled, not run on hardware
//
#include <hip/hip_runtime.h>

// Time-aware full attention for MI455X (gfx1250), flash-attention style.
// decay folded into Q (with log2e so softmax runs in exp2 domain, using the
// raw v_exp_f32 via __builtin_amdgcn_exp2f); scores computed transposed
// (S^T = K x Q^T) so the WMMA D-layout feeds the P operand of the P*V matmul
// with zero data movement. K/V staged through double-buffered LDS (f16) once
// per workgroup with a register-buffered software pipeline: global loads for
// block n+1 are in flight while block n's WMMAs run, and ds stores for block
// n+1 land in the buffer block n-1 used, so one barrier per block suffices.

typedef __attribute__((ext_vector_type(16))) _Float16 v16h;
typedef __attribute__((ext_vector_type(8)))  _Float16 v8h;
typedef __attribute__((ext_vector_type(2)))  _Float16 v2h;
typedef __attribute__((ext_vector_type(8)))  float    v8f;

static __device__ __forceinline__ v8f wmma_f16(v16h a, v16h b, v8f c) {
  // D = A x B + C, 16x16x32, f16 in / f32 accumulate
  return __builtin_amdgcn_wmma_f32_16x16x32_f16(false, a, false, b, (short)0, c,
                                                false, false);
}

static __device__ __forceinline__ v16h cat8(v8h lo, v8h hi) {
  return __builtin_shufflevector(lo, hi, 0, 1, 2, 3, 4, 5, 6, 7,
                                 8, 9, 10, 11, 12, 13, 14, 15);
}

// raw hardware exp2 (v_exp_f32), no OCML denorm-range fixup; args are <= 0
// and values below -126 flush to 0, which is exactly what softmax wants.
static __device__ __forceinline__ float fast_exp2(float x) {
  return __builtin_amdgcn_exp2f(x);
}

namespace {
constexpr int Lq = 1024;
constexpr int Sk = 1024;
constexpr int Ee = 64;
constexpr float FACTOR = 5.0f;
constexpr float SCALE  = 0.4472135954999579f;   // 1/sqrt(5)
constexpr float LOG2E  = 1.4426950408889634f;   // run softmax in exp2 domain
// LDS row strides (in halves), chosen for b128 alignment + bank spread:
constexpr int KR = 72;  // sK row: 144B = 36 dwords -> 16 distinct banks on reads
constexpr int VR = 40;  // sVT row: 80B = 20 dwords -> 16 distinct banks on reads
}

__global__ __launch_bounds__(128) void ta_flash_attn_kernel(
    const float* __restrict__ Q, const float* __restrict__ K,
    const float* __restrict__ V, const float* __restrict__ T,
    float* __restrict__ O) {
  // double-buffered tiles: [buf][key/E-major]
  __shared__ __align__(16) _Float16 sK[2][32 * KR];   // [key 0..31][E 0..63]
  __shared__ __align__(16) _Float16 sVT[2][64 * VR];  // [E 0..63][key 0..31]

  const int tid  = threadIdx.x;
  const int lane = tid & 31;
  const int wave = tid >> 5;
  const int q    = lane & 15;   // query col (B-op / D-layout), key row (A-op)
  const int h2   = lane >> 4;   // lane-half

  const int bh   = blockIdx.y;            // 0..B*H-1
  const int b    = bh >> 3;               // H == 8
  const int row0 = blockIdx.x * 64 + wave * 16;

  const float* Qb = Q + (size_t)bh * Lq * Ee;
  const float* Kb = K + (size_t)bh * Sk * Ee;
  const float* Vb = V + (size_t)bh * Sk * Ee;
  float*       Ob = O + (size_t)bh * Lq * Ee;

  // fold per-row decay + softmax scale + log2(e) into Q
  const float cq = SCALE * LOG2E * __expf(-T[b * Lq + row0 + q] / FACTOR);

  // ---- Q as WMMA B-operand: lane holds query row (row0+q), elem e<8 -> E = 32f + 8*h2 + e, e>=8 -> E = 32f + 16 + 8*h2 + (e-8)
  v16h Qf[2];
  {
    const float* qrow = Qb + (size_t)(row0 + q) * Ee;
#pragma unroll
    for (int f = 0; f < 2; ++f) {
      const float4* p4 = reinterpret_cast<const float4*>(qrow + 32 * f + 8 * h2);
#pragma unroll
      for (int v = 0; v < 4; ++v) {
        float4 x = p4[v + (v & 2)];
        Qf[f][4 * v + 0] = (_Float16)(x.x * cq);
        Qf[f][4 * v + 1] = (_Float16)(x.y * cq);
        Qf[f][4 * v + 2] = (_Float16)(x.z * cq);
        Qf[f][4 * v + 3] = (_Float16)(x.w * cq);
      }
    }
  }

  // ---- staging maps (per thread):
  // K: pass p: row p*16 + krA, cols c0A..c0A+7 -> one v8h ds_store_b128
  // V: pass p: rows p*16 + krB, +1 (pair), cols c0B..c0B+3 -> packed ds_store_b32
  const int krA = (tid >> 3) & 15;       // 0..15
  const int c0A = (tid & 7) * 8;         // 0,8,..,56
  const int krB = 2 * ((tid >> 4) & 7);  // 0,2,..,14
  const int c0B = (tid & 15) * 4;        // 0,4,..,60

  float4 kraw[4], vraw[4];

  auto load_block = [&](int kbb) {
#pragma unroll
    for (int p = 0; p < 2; ++p) {
      const float* kp = Kb + (size_t)(kbb + p * 16 + krA) * Ee + c0A;
      kraw[2 * p + 0] = *reinterpret_cast<const float4*>(kp);
      kraw[2 * p + 1] = *reinterpret_cast<const float4*>(kp + 4);
      const float* vp = Vb + (size_t)(kbb + p * 16 + krB) * Ee + c0B;
      vraw[2 * p + 0] = *reinterpret_cast<const float4*>(vp);
      vraw[2 * p + 1] = *reinterpret_cast<const float4*>(vp + Ee);
    }
  };

  auto store_block = [&](int buf) {
#pragma unroll
    for (int p = 0; p < 2; ++p) {
      const float* a0 = reinterpret_cast<const float*>(&kraw[2 * p + 0]);
      const float* a1 = reinterpret_cast<const float*>(&kraw[2 * p + 1]);
      v8h hk;
#pragma unroll
      for (int j = 0; j < 4; ++j) {
        hk[j]     = (_Float16)a0[j];
        hk[4 + j] = (_Float16)a1[j];
      }
      *reinterpret_cast<v8h*>(&sK[buf][(p * 16 + krA) * KR + c0A]) = hk;

      const float* v0 = reinterpret_cast<const float*>(&vraw[2 * p + 0]);
      const float* v1 = reinterpret_cast<const float*>(&vraw[2 * p + 1]);
#pragma unroll
      for (int j = 0; j < 4; ++j) {
        v2h hv;
        hv[0] = (_Float16)v0[j];
        hv[1] = (_Float16)v1[j];
        *reinterpret_cast<v2h*>(&sVT[buf][(c0B + j) * VR + (p * 16 + krB)]) = hv;
      }
    }
  };

  // online-softmax state (replicated in both lane halves; row = query q)
  float m_i = -1e30f, l_i = 0.0f;
  v8f Oacc[4];
#pragma unroll
  for (int c = 0; c < 4; ++c)
#pragma unroll
    for (int r = 0; r < 8; ++r) Oacc[c][r] = 0.0f;

  // pipeline prologue: stage block 0 into buffer 0
  load_block(0);
  store_block(0);
  __syncthreads();

  int ib = 0;
  for (int kb = 0; kb < Sk; kb += 32, ib ^= 1) {
    const bool more = (kb + 32 < Sk);
    if (more) load_block(kb + 32);  // HBM latency hidden behind this block's WMMAs

    // ---- K as WMMA A-operand from LDS: lane = key row, 8-half runs are one b128 each
    v16h Kf[2][2];
#pragma unroll
    for (int kt = 0; kt < 2; ++kt) {
      const _Float16* kp = &sK[ib][(16 * kt + q) * KR];
#pragma unroll
      for (int f = 0; f < 2; ++f) {
        v8h lo = *reinterpret_cast<const v8h*>(kp + 32 * f + 8 * h2);
        v8h hi = *reinterpret_cast<const v8h*>(kp + 32 * f + 16 + 8 * h2);
        Kf[kt][f] = cat8(lo, hi);
      }
    }

    // ---- S^T = K x Q^T: D-layout -> lane = query q, VGPR r = key (kt*16 + 8*h2 + r)
    v8f S0, S1;
#pragma unroll
    for (int r = 0; r < 8; ++r) { S0[r] = 0.0f; S1[r] = 0.0f; }
    S0 = wmma_f16(Kf[0][0], Qf[0], S0);
    S0 = wmma_f16(Kf[0][1], Qf[1], S0);
    S1 = wmma_f16(Kf[1][0], Qf[0], S1);
    S1 = wmma_f16(Kf[1][1], Qf[1], S1);

    // ---- online softmax in exp2 domain (lane sees 16 of 32 keys; rest in lane^16)
    float a8[8];
#pragma unroll
    for (int r = 0; r < 8; ++r) a8[r] = fmaxf(S0[r], S1[r]);
    float b0 = fmaxf(a8[0], a8[1]), b1 = fmaxf(a8[2], a8[3]);
    float b2 = fmaxf(a8[4], a8[5]), b3 = fmaxf(a8[6], a8[7]);
    float bm = fmaxf(fmaxf(b0, b1), fmaxf(b2, b3));
    bm = fmaxf(bm, __shfl_xor(bm, 16, 32));
    const float m_new = fmaxf(m_i, bm);
    const float alpha = fast_exp2(m_i - m_new);

    v16h Pf;   // A-operand of P*V: e<8 -> S0[e], e>=8 -> S1[e-8]
    float ps[8];
#pragma unroll
    for (int r = 0; r < 8; ++r) {
      float p0 = fast_exp2(S0[r] - m_new);
      float p1 = fast_exp2(S1[r] - m_new);
      Pf[r]     = (_Float16)p0;
      Pf[8 + r] = (_Float16)p1;
      ps[r] = p0 + p1;
    }
    float t0 = ps[0] + ps[1], t1 = ps[2] + ps[3];
    float t2 = ps[4] + ps[5], t3 = ps[6] + ps[7];
    float ls = (t0 + t1) + (t2 + t3);
    ls += __shfl_xor(ls, 16, 32);
    l_i = l_i * alpha + ls;
    m_i = m_new;

    // per-output-row (m = r + 8*h2) rescale factors from the lane=query layout
    float al[8];
#pragma unroll
    for (int r = 0; r < 8; ++r) al[r] = __shfl(alpha, r + 8 * h2, 32);

    // ---- O = alpha*O + P x V ; V B-operand = two 8-half runs of sVT column (keys 8*h2.., 16+8*h2..)
#pragma unroll
  for (int c = 0; c < 4; ++c) {
      const _Float16* vp = &sVT[ib][(16 * c + q) * VR + 8 * h2];
      v8h lo = *reinterpret_cast<const v8h*>(vp);
      v8h hi = *reinterpret_cast<const v8h*>(vp + 16);
      v16h Vf = cat8(lo, hi);
#pragma unroll
      for (int r = 0; r < 8; ++r) Oacc[c][r] *= al[r];
      Oacc[c] = wmma_f16(Pf, Vf, Oacc[c]);
    }

    // stores go to the buffer that was read in iteration n-1; the single
    // end-of-iteration barrier both released that buffer and will publish
    // these stores before iteration n+1 reads them.
    if (more) store_block(ib ^ 1);
    __syncthreads();
  }

  // ---- finalize: divide by row sums, write out (lane = e-col, VGPR r = row)
  const float rl = 1.0f / l_i;
  float inv[8];
#pragma unroll
  for (int r = 0; r < 8; ++r) inv[r] = __shfl(rl, r + 8 * h2, 32);

#pragma unroll
  for (int c = 0; c < 4; ++c)
#pragma unroll
    for (int r = 0; r < 8; ++r)
      Ob[(size_t)(row0 + r + 8 * h2) * Ee + 16 * c + q] = Oacc[c][r] * inv[r];
}

extern "C" void kernel_launch(void* const* d_in, const int* in_sizes, int n_in,
                              void* d_out, int out_size, void* d_ws, size_t ws_size,
                              hipStream_t stream) {
  (void)in_sizes; (void)n_in; (void)d_ws; (void)ws_size; (void)out_size;
  const float* Q = (const float*)d_in[0];
  const float* K = (const float*)d_in[1];
  const float* V = (const float*)d_in[2];
  const float* T = (const float*)d_in[3];
  float* O = (float*)d_out;

  dim3 grid(Lq / 64, /*B*H=*/64, 1);
  dim3 block(128, 1, 1);
  ta_flash_attn_kernel<<<grid, block, 0, stream>>>(Q, K, V, T, O);
}
